// MyLegacyLSTM_72069551227227
// MI455X (gfx1250) — matrix operand order repaired and checked by reading, not compiled, not run on hardware
//
#include <hip/hip_runtime.h>

// ---------------------------------------------------------------------------
// LSTM for MI455X (gfx1250, wave32, WMMA).
//   Phase A: X_g = x @ W_g + b_g  (4 gates) via v_wmma_f32_16x16x32_bf16.
//   Phase B: persistent 16-block scan over T=2048 steps; U_g resident in LDS
//            (bf16, N-partitioned), c in f32 LDS, h exchanged via L2 (bf16),
//            one agent-scope sense-reversing grid barrier per step.
//            h reload uses GLOBAL_LOAD_ASYNC_TO_LDS (ASYNCcnt) when available;
//            X(t+1) prefetched (global_prefetch_b8) before the barrier spin.
// ---------------------------------------------------------------------------

typedef unsigned short u16;
typedef unsigned int   u32;
typedef __attribute__((ext_vector_type(16))) __bf16 v16bf;
typedef __attribute__((ext_vector_type(8)))  float  v8f;

#ifndef __has_builtin
#define __has_builtin(x) 0
#endif
#if __has_builtin(__builtin_amdgcn_global_load_async_to_lds_b128) && \
    __has_builtin(__builtin_amdgcn_s_wait_asynccnt)
#define USE_ASYNC_LDS 1
#else
#define USE_ASYNC_LDS 0
#endif

// Trivial 16-byte chunk (HIP's uint4 has non-trivial ctor -> not union-safe).
struct __attribute__((aligned(16))) B128 { u32 a, b, c, d; };
union FragU { v16bf v; B128 q[2]; };

#if USE_ASYNC_LDS
// Builtin signature (from hipcc diagnostic): param 0 is
// 'int __attribute__((vector_size(16))) __device__ *' (global), param 1 LDS.
typedef int async_v4i __attribute__((vector_size(16)));
static __device__ __forceinline__ void async_copy_b128(const void* g, void* l) {
  __builtin_amdgcn_global_load_async_to_lds_b128(
      (__attribute__((address_space(1))) async_v4i*)(g),
      (__attribute__((address_space(3))) async_v4i*)(l),
      0, 0);
}
#endif

static __device__ __forceinline__ u16 f32_to_bf16(float f) {
  u32 u = __float_as_uint(f);
  u32 r = 0x7FFFu + ((u >> 16) & 1u);   // round-to-nearest-even
  return (u16)((u + r) >> 16);
}

static __device__ __forceinline__ float fast_tanh(float x) {
#if __has_builtin(__builtin_amdgcn_tanh_f32)
  return __builtin_amdgcn_tanh_f32(x);       // v_tanh_f32 (TRANS32)
#else
  return tanhf(x);
#endif
}

static __device__ __forceinline__ float fast_sigmoid(float x) {
  return __builtin_amdgcn_rcpf(1.0f + __expf(-x));  // v_rcp_f32 / v_exp_f32
}

// A-matrix 16x32 bf16 fragment (ISA 7.12.2): lane L holds row M=L%16;
// lanes 0-15: K = k0+0..7 (V0-3), k0+16..23 (V4-7); lanes 16-31: +8.
static __device__ __forceinline__ v16bf load_a_frag(const u16* base, int ldr,
                                                    int row0, int k0, int lane) {
  const u16* p = base + (size_t)(row0 + (lane & 15)) * ldr + k0 + ((lane >> 4) << 3);
  FragU f;
  f.q[0] = *(const B128*)(p);
  f.q[1] = *(const B128*)(p + 16);
  return f.v;
}

// B-matrix 32x16 bf16 fragment from an [N][K] (pre-transposed) tile:
// lanes 0-15: N = n0+lane, K = k0+0..7, k0+16..23; lanes 16-31: N = n0+lane-16, +8.
static __device__ __forceinline__ v16bf load_b_frag(const u16* base, int ldr,
                                                    int n0, int k0, int lane) {
  const u16* p = base + (size_t)(n0 + (lane & 15)) * ldr + k0 + ((lane >> 4) << 3);
  FragU f;
  f.q[0] = *(const B128*)(p);
  f.q[1] = *(const B128*)(p + 16);
  return f.v;
}

static __device__ __forceinline__ v8f wmma_bf16(v16bf a, v16bf b, v8f c) {
  // (neg_a, A, neg_b, B, c_mod, C, reuse_a, reuse_b)
  return __builtin_amdgcn_wmma_f32_16x16x32_bf16(false, a, false, b, (short)0, c,
                                                 false, false);
}

// ---------------------------------------------------------------------------
// Init: zero grid-barrier words and the h exchange double buffer.
__global__ void __launch_bounds__(256) lstm_init_kernel(u32* bar, B128* hbuf4) {
  int tid = threadIdx.x;
  if (tid < 16) bar[tid] = 0u;
  B128 z = {0u, 0u, 0u, 0u};
  for (int i = tid; i < 2048; i += 256) hbuf4[i] = z;   // 2*32*256 bf16 = 32KB
}

// Convert x (f32) -> bf16, 4 elements/thread.
__global__ void __launch_bounds__(256) lstm_cvt_x_kernel(const float* __restrict__ x,
                                                         u16* __restrict__ xb, int n4) {
  int idx = blockIdx.x * 256 + threadIdx.x;
  if (idx >= n4) return;
  float4 v = ((const float4*)x)[idx];
  u32 lo = (u32)f32_to_bf16(v.x) | ((u32)f32_to_bf16(v.y) << 16);
  u32 hi = (u32)f32_to_bf16(v.z) | ((u32)f32_to_bf16(v.w) << 16);
  ((uint2*)xb)[idx] = make_uint2(lo, hi);
}

// Convert + transpose W_g, U_g (256x256 f32 [K][N]) -> bf16 [N][K].
__global__ void __launch_bounds__(256) lstm_cvt_w_kernel(
    const float* w0, const float* w1, const float* w2, const float* w3,
    const float* u0, const float* u1, const float* u2, const float* u3,
    u16* __restrict__ wT, u16* __restrict__ uT) {
  int idx = blockIdx.x * 256 + threadIdx.x;   // 0 .. 8*65536-1
  int m = idx >> 16;
  int r = idx & 65535;
  int n = r >> 8, k = r & 255;
  const float* src = (m == 0) ? w0 : (m == 1) ? w1 : (m == 2) ? w2 : (m == 3) ? w3
                   : (m == 4) ? u0 : (m == 5) ? u1 : (m == 6) ? u2 : u3;
  u16 v = f32_to_bf16(src[k * 256 + n]);
  if (m < 4) wT[(size_t)m * 65536 + (size_t)n * 256 + k] = v;
  else       uT[(size_t)(m - 4) * 65536 + (size_t)n * 256 + k] = v;
}

// ---------------------------------------------------------------------------
// Phase A: X[g][t][b][:] = x[b][t][:] @ W_g + b_g   (time-major for the scan).
// Block = 64x64 output tile, 8 waves, each wave: 1 m-tile x 2 n-tiles, K=256.
__global__ void __launch_bounds__(256) lstm_phaseA_kernel(
    const u16* __restrict__ xb, const u16* __restrict__ wT,
    const float* __restrict__ bF, const float* __restrict__ bI,
    const float* __restrict__ bO, const float* __restrict__ bC,
    float* __restrict__ X) {
  __shared__ u16 sA[64][256];   // 32KB: x rows m0..m0+63, K=256
  __shared__ u16 sB[64][256];   // 32KB: wT rows n0..n0+63, K=256
  const int tid = threadIdx.x;
  const int n0 = blockIdx.x * 64;
  const int m0 = blockIdx.y * 64;
  const int g  = blockIdx.z;

  {
    int row = tid >> 2;     // 0..63
    int c4  = tid & 3;      // 8 x uint4 each
    const uint4* srcA = (const uint4*)(xb + (size_t)(m0 + row) * 256);
    const uint4* srcB = (const uint4*)(wT + ((size_t)g * 256 + n0 + row) * 256);
    uint4* dA = (uint4*)&sA[row][0];
    uint4* dB = (uint4*)&sB[row][0];
#pragma unroll
    for (int i = 0; i < 8; ++i) {
      dA[c4 * 8 + i] = srcA[c4 * 8 + i];
      dB[c4 * 8 + i] = srcB[c4 * 8 + i];
    }
  }
  __syncthreads();

  const int wave = tid >> 5, lane = tid & 31;
  const int mt  = wave & 3;
  const int nt0 = (wave >> 2) * 2;
  v8f acc0{}; v8f acc1{};
#pragma unroll
  for (int k0 = 0; k0 < 256; k0 += 32) {
    v16bf a  = load_a_frag(&sA[0][0], 256, mt * 16, k0, lane);
    v16bf b0 = load_b_frag(&sB[0][0], 256, nt0 * 16, k0, lane);
    v16bf b1 = load_b_frag(&sB[0][0], 256, (nt0 + 1) * 16, k0, lane);
    acc0 = wmma_bf16(a, b0, acc0);
    acc1 = wmma_bf16(a, b1, acc1);
  }
  const float* bias = (g == 0) ? bF : (g == 1) ? bI : (g == 2) ? bO : bC;
  const int nc0 = n0 + nt0 * 16 + (lane & 15);
  const float bv0 = bias[nc0], bv1 = bias[nc0 + 16];
#pragma unroll
  for (int r = 0; r < 8; ++r) {
    int m = m0 + mt * 16 + ((lane < 16) ? r : (r + 8));
    int b = m >> 11;            // m = b*T + t, T = 2048
    int t = m & 2047;
    size_t off = (((size_t)g * 2048 + t) * 32 + b) * 256;
    X[off + nc0]      = acc0[r] + bv0;
    X[off + nc0 + 16] = acc1[r] + bv1;
  }
}

// ---------------------------------------------------------------------------
// Agent-scope sense-reversing grid barrier (16 persistent blocks).
static __device__ __forceinline__ void grid_barrier(u32* bar, u32 nb) {
  __threadfence();          // release block's global stores (agent scope)
  __syncthreads();
  if (threadIdx.x == 0) {
    u32 gen  = __hip_atomic_load(bar + 1, __ATOMIC_RELAXED, __HIP_MEMORY_SCOPE_AGENT);
    u32 prev = __hip_atomic_fetch_add(bar, 1u, __ATOMIC_ACQ_REL, __HIP_MEMORY_SCOPE_AGENT);
    if (prev == nb - 1) {
      __hip_atomic_store(bar, 0u, __ATOMIC_RELAXED, __HIP_MEMORY_SCOPE_AGENT);
      __hip_atomic_fetch_add(bar + 1, 1u, __ATOMIC_RELEASE, __HIP_MEMORY_SCOPE_AGENT);
    } else {
      while (__hip_atomic_load(bar + 1, __ATOMIC_ACQUIRE, __HIP_MEMORY_SCOPE_AGENT) == gen)
        __builtin_amdgcn_s_sleep(2);
    }
  }
  __syncthreads();
  __builtin_amdgcn_fence(__ATOMIC_ACQUIRE, "agent");  // invalidate L0 for h reload
}

// Phase B: sequential scan. 16 blocks x 128 threads; block owns H-columns
// [16*wg, 16*wg+16) for all 4 gates. Wave g computes pre_g = h @ U_g slice.
__global__ void __launch_bounds__(128) lstm_phaseB_kernel(
    const float* __restrict__ X, const u16* __restrict__ uT,
    u16* __restrict__ hbuf, u32* __restrict__ bar, float* __restrict__ out) {
  __shared__ u16   sU[4][16][256];   // 32KB  U_g^T slices [n_local][k]
  __shared__ u16   sH[32][256];      // 16KB  h (bf16), rows = batch
  __shared__ float sPre[4][32][16];  // 8KB   h@U per gate
  __shared__ float sC[32][16];       // 2KB   cell state (exact f32)
  const int tid = threadIdx.x;
  const int n0  = blockIdx.x * 16;

  for (int i = tid; i < 2048; i += 128) {          // 4*16 rows x 32 uint4
    int gn = i >> 5, c = i & 31;
    int gg = gn >> 4, nn = gn & 15;
    ((uint4*)&sU[gg][nn][0])[c] =
        ((const uint4*)(uT + ((size_t)gg * 256 + n0 + nn) * 256))[c];
  }
  for (int i = tid; i < 1024; i += 128) ((uint4*)&sH[0][0])[i] = make_uint4(0, 0, 0, 0);
  for (int i = tid; i < 512;  i += 128) (&sC[0][0])[i] = 0.0f;
  __syncthreads();

  const int wave = tid >> 5, lane = tid & 31;
  const int g   = wave;           // one gate per wave
  const int eb  = tid >> 2;       // elementwise: batch row 0..31
  const int ej0 = (tid & 3) * 4;  // elementwise: 4 columns
  float hlast[4] = {0.f, 0.f, 0.f, 0.f};

  for (int t = 0; t < 2048; ++t) {
    // Issue X(t) loads early so L2 latency hides under the GEMM.
    float4 xg[4];
#pragma unroll
    for (int gg = 0; gg < 4; ++gg)
      xg[gg] = *(const float4*)(X + (((size_t)gg * 2048 + t) * 32 + eb) * 256 + n0 + ej0);

    // h(32x256) @ U_g-slice(256x16): 2 m-tiles x 8 k-steps = 16 WMMAs/wave.
    v8f acc0{}; v8f acc1{};
#pragma unroll
    for (int k0 = 0; k0 < 256; k0 += 32) {
      v16bf b  = load_b_frag(&sU[g][0][0], 256, 0, k0, lane);
      v16bf a0 = load_a_frag(&sH[0][0], 256, 0, k0, lane);
      v16bf a1 = load_a_frag(&sH[0][0], 256, 16, k0, lane);
      acc0 = wmma_bf16(a0, b, acc0);
      acc1 = wmma_bf16(a1, b, acc1);
    }
#pragma unroll
    for (int r = 0; r < 8; ++r) {
      int rr = (lane < 16) ? r : (r + 8);
      sPre[g][rr][lane & 15]      = acc0[r];
      sPre[g][16 + rr][lane & 15] = acc1[r];
    }
    __syncthreads();

    // Elementwise gate math; each thread owns (eb, ej0..ej0+3).
    float hnew[4]; u16 hb[4];
#pragma unroll
    for (int jj = 0; jj < 4; ++jj) {
      int j = ej0 + jj;
      float pf = sPre[0][eb][j] + ((const float*)&xg[0])[jj];
      float pi = sPre[1][eb][j] + ((const float*)&xg[1])[jj];
      float po = sPre[2][eb][j] + ((const float*)&xg[2])[jj];
      float pc = sPre[3][eb][j] + ((const float*)&xg[3])[jj];
      float f = fast_sigmoid(pf);
      float i = fast_sigmoid(pi);
      float o = fast_sigmoid(po);
      float cn = f * sC[eb][j] + i * fast_tanh(pc);
      sC[eb][j] = cn;
      float h = o * fast_tanh(cn);
      hnew[jj] = h;
      hlast[jj] = h;
      hb[jj] = f32_to_bf16(h);
    }
    // y[b][t][:] = h(t+1)
    *(float4*)(out + ((size_t)eb * 2048 + t) * 256 + n0 + ej0) =
        make_float4(hnew[0], hnew[1], hnew[2], hnew[3]);
    // publish h (bf16) to double buffer parity (t+1)&1
    {
      u32 lo = (u32)hb[0] | ((u32)hb[1] << 16);
      u32 hi = (u32)hb[2] | ((u32)hb[3] << 16);
      *(uint2*)(hbuf + (size_t)((t + 1) & 1) * 32 * 256 + (size_t)eb * 256 + n0 + ej0) =
          make_uint2(lo, hi);
    }

    if (t != 2047) {
      // Prefetch next step's X slices into cache while we wait at the barrier.
#pragma unroll
      for (int gg = 0; gg < 4; ++gg)
        __builtin_prefetch(
            X + (((size_t)gg * 2048 + (t + 1)) * 32 + eb) * 256 + n0 + ej0, 0, 3);

      grid_barrier(bar, 16);

      // Reload full h(t+1) (16KB bf16) from the L2-resident exchange buffer.
#if USE_ASYNC_LDS
      {
        const char* srcg = (const char*)(hbuf + (size_t)((t + 1) & 1) * 32 * 256);
        char* dstl = (char*)&sH[0][0];
        for (int i = tid; i < 1024; i += 128)
          async_copy_b128(srcg + (size_t)i * 16, dstl + (size_t)i * 16);
        __builtin_amdgcn_s_wait_asynccnt(0);
      }
#else
      {
        const uint4* src = (const uint4*)(hbuf + (size_t)((t + 1) & 1) * 32 * 256);
        for (int i = tid; i < 1024; i += 128) ((uint4*)&sH[0][0])[i] = src[i];
      }
#endif
      __syncthreads();
    }
  }

  // Final (h, c) after y: out[BTH .. BTH+BH) = h, then c.
  size_t base = (size_t)32 * 2048 * 256;
  *(float4*)(out + base + (size_t)eb * 256 + n0 + ej0) =
      make_float4(hlast[0], hlast[1], hlast[2], hlast[3]);
  *(float4*)(out + base + 32 * 256 + (size_t)eb * 256 + n0 + ej0) =
      make_float4(sC[eb][ej0], sC[eb][ej0 + 1], sC[eb][ej0 + 2], sC[eb][ej0 + 3]);
}

// ---------------------------------------------------------------------------
extern "C" void kernel_launch(void* const* d_in, const int* in_sizes, int n_in,
                              void* d_out, int out_size, void* d_ws, size_t ws_size,
                              hipStream_t stream) {
  (void)in_sizes; (void)n_in; (void)out_size; (void)ws_size;
  const float* x  = (const float*)d_in[0];
  const float* wf = (const float*)d_in[1];
  const float* uf = (const float*)d_in[2];
  const float* bf = (const float*)d_in[3];
  const float* wi = (const float*)d_in[4];
  const float* ui = (const float*)d_in[5];
  const float* bi = (const float*)d_in[6];
  const float* wo = (const float*)d_in[7];
  const float* uo = (const float*)d_in[8];
  const float* bo = (const float*)d_in[9];
  const float* wc = (const float*)d_in[10];
  const float* uc = (const float*)d_in[11];
  const float* bc = (const float*)d_in[12];

  // Workspace layout (16B-aligned): bar | hbuf | x_bf16 | wT | uT | X(f32)
  char* ws   = (char*)d_ws;
  u32*  bar  = (u32*)ws;                                   // 256 B
  u16*  hbuf = (u16*)(ws + 256);                           // 32 KB
  u16*  xb   = (u16*)(ws + 256 + 32768);                   // 32 MB
  u16*  wT   = xb + (size_t)32 * 2048 * 256;               // 512 KB
  u16*  uT   = wT + (size_t)4 * 65536;                     // 512 KB
  float* X   = (float*)(uT + (size_t)4 * 65536);           // 268 MB

  lstm_init_kernel<<<1, 256, 0, stream>>>(bar, (B128*)hbuf);
  lstm_cvt_x_kernel<<<(16777216 / 4) / 256, 256, 0, stream>>>(x, xb, 16777216 / 4);
  lstm_cvt_w_kernel<<<2048, 256, 0, stream>>>(wf, wi, wo, wc, uf, ui, uo, uc, wT, uT);
  lstm_phaseA_kernel<<<dim3(4, 1024, 4), 256, 0, stream>>>(xb, wT, bf, bi, bo, bc, X);
  lstm_phaseB_kernel<<<16, 128, 0, stream>>>(X, uT, hbuf, bar, (float*)d_out);
}
